// ItemEncoder_30674656428498
// MI455X (gfx1250) — compile-verified, matrix operand order checked by reading, not run on hardware
//
#include <hip/hip_runtime.h>

#define IN_DIM  64
#define HID_DIM 256
#define OUT_DIM 128

typedef __attribute__((ext_vector_type(16))) __bf16       v16bf;
typedef __attribute__((ext_vector_type(8)))  float        v8f;
typedef __attribute__((ext_vector_type(8)))  unsigned int v8u;

// RNE f32->bf16 (used only for one-time weight staging)
static __device__ __forceinline__ unsigned short f2bf_rne(float f) {
  unsigned int u = __builtin_bit_cast(unsigned int, f);
  u += 0x7FFFu + ((u >> 16) & 1u);
  return (unsigned short)(u >> 16);
}
static __device__ __forceinline__ unsigned int packbf_rne(float lo, float hi) {
  return (unsigned int)f2bf_rne(lo) | ((unsigned int)f2bf_rne(hi) << 16);
}
// Truncating pack {bf16(hi), bf16(lo)} in one v_perm_b32 (activation path)
static __device__ __forceinline__ unsigned int packbf_trunc(float lo, float hi) {
  return __builtin_amdgcn_perm(__builtin_bit_cast(unsigned int, hi),
                               __builtin_bit_cast(unsigned int, lo),
                               0x07060302u);
}
// ReLU in one v_med3_f32
static __device__ __forceinline__ float relu1(float v) {
  return __builtin_amdgcn_fmed3f(v, 0.0f, 3.402823466e+38f);
}
// K index held by (lane-half h, dword v) for 16-bit A/B operands (ISA 7.12.2)
static __device__ __forceinline__ int kmapf(int h, int v) {
  return (v < 4) ? (8*h + 2*v) : (16 + 8*h + 2*(v-4));
}

__global__ void zero_kernel(float* __restrict__ p, int n) {
  int i = blockIdx.x * blockDim.x + threadIdx.x;
  if (i < n) p[i] = 0.0f;
}

__global__ __launch_bounds__(256)
void item_encoder_kernel(const float* __restrict__ x,
                         const long long* __restrict__ idxs,
                         const float* __restrict__ W1,
                         const float* __restrict__ b1,
                         const float* __restrict__ W2,
                         const float* __restrict__ b2,
                         float* __restrict__ out,
                         int nItems)
{
  // W1: 16 N-tiles x 2 K-chunks, 1KB blocks in WMMA-B layout (32KB)
  // W2: 8 K-chunks x 8 N-tiles, 1KB blocks in WMMA-B layout   (64KB)
  // A2: per-wave 2x1KB bounce buffers (two M-tiles), C/D layout -> A layout
  __shared__ __align__(32) unsigned int  ldsW1[IN_DIM*HID_DIM/2];
  __shared__ __align__(32) unsigned int  ldsW2[HID_DIM*OUT_DIM/2];
  __shared__ __align__(32) unsigned char ldsA2[8 * 2048];
  __shared__ float ldsB1[HID_DIM];
  __shared__ float ldsB2[OUT_DIM];

  const int tid = threadIdx.x;

  // ---- stage W1 (f32 -> bf16 RNE, pre-swizzled into B-operand layout) ----
  for (int g = tid; g < IN_DIM*HID_DIM/2; g += 256) {
    int b  = g >> 8;                 // block = nt*2 + kc
    int nt = b >> 1, kc = b & 1;
    int l  = (g >> 3) & 31, v = g & 7;
    int h  = l >> 4;
    int col = nt*16 + (l & 15);
    int k  = kc*32 + kmapf(h, v);
    ldsW1[g] = packbf_rne(W1[k*HID_DIM + col], W1[(k+1)*HID_DIM + col]);
  }
  // ---- stage W2 ----
  for (int g = tid; g < HID_DIM*OUT_DIM/2; g += 256) {
    int b  = g >> 8;                 // block = kc*8 + nt
    int kc = b >> 3, nt = b & 7;
    int l  = (g >> 3) & 31, v = g & 7;
    int h  = l >> 4;
    int col = nt*16 + (l & 15);
    int k  = kc*32 + kmapf(h, v);
    ldsW2[g] = packbf_rne(W2[k*OUT_DIM + col], W2[(k+1)*OUT_DIM + col]);
  }
  // ---- stage biases (avoid pinning 24 VGPRs for the whole kernel) ----
  if (tid < HID_DIM) ldsB1[tid] = b1[tid];
  if (tid < OUT_DIM) ldsB2[tid] = b2[tid];
  __syncthreads();

  const int wave = tid >> 5;         // wave32
  const int lane = tid & 31;
  const int hl   = lane >> 4;        // lane half
  const int ln   = lane & 15;
  unsigned short* a2s[2];
  const unsigned char* a2rd[2];
  #pragma unroll
  for (int m = 0; m < 2; ++m) {
    a2s[m]  = (unsigned short*)(ldsA2 + wave*2048 + m*1024);
    a2rd[m] = ldsA2 + wave*2048 + m*1024 + lane*32;
  }

  const int nTiles = (nItems + 255) >> 8;        // 256 items per block-tile
  for (int tile = blockIdx.x; tile < nTiles; tile += gridDim.x) {
    { // prefetch next tile of x into caches (256B per thread covers 64KB tile)
      long long nrow = (long long)(tile + gridDim.x) * 256;
      if (nrow < nItems)
        __builtin_prefetch(x + nrow*IN_DIM + (long long)tid*64, 0, 0);
    }
    const int rowbase = tile*256 + wave*32;      // this wave's 32 rows

    // ---- A operands from x: 2 M-tiles x 2 K32 chunks, truncate-pack ----
    v8u A[2][2];
    #pragma unroll
    for (int m = 0; m < 2; ++m) {
      const int xr = rowbase + m*16 + ln;
      const float* xrow = x + (size_t)((xr < nItems) ? xr : 0) * IN_DIM;
      #pragma unroll
      for (int c = 0; c < 2; ++c) {
        const float* r0 = xrow + c*32 + 8*hl;    // K = 8h .. 8h+7
        const float* r1 = r0 + 16;               // K = 16+8h .. 16+8h+7
        float4 q0 = *(const float4*)(r0);
        float4 q1 = *(const float4*)(r0 + 4);
        float4 q2 = *(const float4*)(r1);
        float4 q3 = *(const float4*)(r1 + 4);
        A[m][c][0] = packbf_trunc(q0.x, q0.y); A[m][c][1] = packbf_trunc(q0.z, q0.w);
        A[m][c][2] = packbf_trunc(q1.x, q1.y); A[m][c][3] = packbf_trunc(q1.z, q1.w);
        A[m][c][4] = packbf_trunc(q2.x, q2.y); A[m][c][5] = packbf_trunc(q2.z, q2.w);
        A[m][c][6] = packbf_trunc(q3.x, q3.y); A[m][c][7] = packbf_trunc(q3.z, q3.w);
      }
    }

    // ---- out accumulators (2 x 16x128 f32), init with b2 ----
    v8f acc[2][8];
    #pragma unroll
    for (int nt = 0; nt < 8; ++nt) {
      float bv = ldsB2[nt*16 + ln];
      v8f cv = {bv,bv,bv,bv,bv,bv,bv,bv};
      acc[0][nt] = cv;
      acc[1][nt] = cv;
    }

    // ---- fused MLP: stream h in 32-col chunks straight into GEMM2 ----
    #pragma unroll
    for (int ch = 0; ch < 8; ++ch) {
      #pragma unroll
      for (int t = 0; t < 2; ++t) {              // two 16-col tiles of h
        int nt1 = ch*2 + t;
        float bv = ldsB1[nt1*16 + ln];
        v8f c0[2];
        #pragma unroll
        for (int m = 0; m < 2; ++m) { v8f cv = {bv,bv,bv,bv,bv,bv,bv,bv}; c0[m] = cv; }
        #pragma unroll
        for (int kc = 0; kc < 2; ++kc) {         // K=64 over x
          v16bf bm = __builtin_bit_cast(v16bf,
              *(const v8u*)&ldsW1[(nt1*2 + kc)*256 + lane*8]);
          #pragma unroll
          for (int m = 0; m < 2; ++m) {          // B reused for both M-tiles
            v16bf a = __builtin_bit_cast(v16bf, A[m][kc]);
            c0[m] = __builtin_amdgcn_wmma_f32_16x16x32_bf16(
                        false, a, false, bm, (short)0, c0[m], false, false);
          }
        }
        // relu (v_med3) + truncating bf16 store into per-wave A2 bounce
        int kq = t*16 + ln;                      // k within this 32-chunk
        int hp = (kq >> 3) & 1;
        int vp = (kq < 16) ? ((kq & 7) >> 1) : (4 + ((kq & 7) >> 1));
        #pragma unroll
        for (int m = 0; m < 2; ++m)
          #pragma unroll
          for (int r = 0; r < 8; ++r) {
            unsigned int u = __builtin_bit_cast(unsigned int, relu1(c0[m][r]));
            int la = (r + 8*hl) + 16*hp;         // destination lane
            a2s[m][la*16 + vp*2 + (kq & 1)] = (unsigned short)(u >> 16);
          }
      }
      asm volatile("s_wait_dscnt 0" ::: "memory");   // wave-local DS ordering
      v16bf a2[2];
      a2[0] = __builtin_bit_cast(v16bf, *(const v8u*)a2rd[0]);
      a2[1] = __builtin_bit_cast(v16bf, *(const v8u*)a2rd[1]);
      #pragma unroll
      for (int nt = 0; nt < 8; ++nt) {           // GEMM2: K-chunk = ch
        v16bf bm = __builtin_bit_cast(v16bf,
            *(const v8u*)&ldsW2[(ch*8 + nt)*256 + lane*8]);
        #pragma unroll
        for (int m = 0; m < 2; ++m) {            // B reused for both M-tiles
          acc[m][nt] = __builtin_amdgcn_wmma_f32_16x16x32_bf16(
                           false, a2[m], false, bm, (short)0, acc[m][nt], false, false);
        }
      }
    }

    // ---- scatter-add into bins (L2-resident 2MB table) ----
    #pragma unroll
    for (int m = 0; m < 2; ++m)
      #pragma unroll
      for (int r = 0; r < 8; ++r) {
        int grow = rowbase + m*16 + r + 8*hl;    // row M = r + 8*hl
        if (grow < nItems) {
          long long bin = idxs[grow];
          float* dst = out + (size_t)bin * OUT_DIM + ln;
          #pragma unroll
          for (int nt = 0; nt < 8; ++nt) {
            unsafeAtomicAdd(dst + nt*16, acc[m][nt][r]);
          }
        }
      }

    // Stop the compiler from software-pipelining next tile's x loads across
    // this whole body (that costs ~64 VGPRs and caused scratch spills).
    // Next-tile latency is covered by the prefetch + other waves.
    asm volatile("" ::: "memory");
  }
}

extern "C" void kernel_launch(void* const* d_in, const int* in_sizes, int n_in,
                              void* d_out, int out_size, void* d_ws, size_t ws_size,
                              hipStream_t stream) {
  const float*     x    = (const float*)d_in[0];
  const long long* idxs = (const long long*)d_in[1];
  // d_in[2] = n_bins scalar (4096, compile-time constant here)
  const float* W1 = (const float*)d_in[3];
  const float* b1 = (const float*)d_in[4];
  const float* W2 = (const float*)d_in[5];
  const float* b2 = (const float*)d_in[6];
  float* out = (float*)d_out;

  const int nItems = in_sizes[0] / IN_DIM;

  zero_kernel<<<(out_size + 255) / 256, 256, 0, stream>>>(out, out_size);

  int nTiles = (nItems + 255) >> 8;
  int blocks = nTiles < 1024 ? nTiles : 1024;    // grid-stride, amortize weight staging
  item_encoder_kernel<<<blocks, 256, 0, stream>>>(x, idxs, W1, b1, W2, b2, out, nItems);
}
